// MHSA_33036888441257
// MI455X (gfx1250) — compile-verified, matrix operand order checked by reading, not run on hardware
//
#include <hip/hip_runtime.h>

// MHSA for MI455X (gfx1250, wave32, WMMA, async global->LDS staging).
// Kernel 1: per-head QKV projection (fp32 -> f16 workspace) via
//           v_wmma_f32_16x16x32_f16. Q is pre-scaled by 1/sqrt(DH); V is
//           stored transposed [BH][DH][S] so kernel 2's gathers are all
//           contiguous 16-byte loads.
// Kernel 2: flash attention, double-buffered K/V^T tiles staged with
//           global_load_async_to_lds_b128 (ASYNCcnt), fp32 online softmax,
//           scores and P*V via WMMA, fp32 output.
// Workspace: 3 * B*H*S*DH f16 = ~25 MB.

typedef __attribute__((ext_vector_type(16))) _Float16 v16h;
typedef __attribute__((ext_vector_type(8)))  _Float16 v8h;
typedef __attribute__((ext_vector_type(8)))  float    v8f;

constexpr int kB = 4, kS = 2048, kD = 512, kH = 8, kDH = 64;

// 16-bit A-operand 16x32 layout (ISA 7.12.2): lane l -> row = l&15, g = l>>4.
//   j in [0,8):  K = g*8 + j          (VGPRs 0-3)
//   j in [8,16): K = 16 + g*8 + (j-8) (VGPRs 4-7)
// B-operand mirrored with N = l&15. Both halves are 8 contiguous K values,
// i.e. one 16-byte load each.
__device__ __forceinline__ int a_kmap(int j, int g) {
  return (j < 8) ? (g * 8 + j) : (8 + g * 8 + j);
}

// Build a v16h fragment from two contiguous 8-element chunks.
__device__ __forceinline__ v16h frag(const v8h lo, const v8h hi) {
  return __builtin_shufflevector(lo, hi, 0, 1, 2, 3, 4, 5, 6, 7,
                                 8, 9, 10, 11, 12, 13, 14, 15);
}

// Async global -> LDS copy, 16 bytes per lane (ASYNCcnt-tracked).
__device__ __forceinline__ void async_b128(unsigned int lds_byte_off,
                                           const void* sbase,
                                           unsigned int voff_bytes) {
  asm volatile("global_load_async_to_lds_b128 %0, %1, %2"
               :: "v"(lds_byte_off), "v"(voff_bytes), "s"(sbase)
               : "memory");
}
__device__ __forceinline__ void wait_async0() {
  asm volatile("s_wait_asynccnt 0" ::: "memory");
}

// ---------------------------------------------------------------------------
// Kernel 1: Q/K/V = X @ W^T + b (per head). Q pre-scaled by 1/8. V transposed.
// Block: 256 threads (8 waves), 128 rows of one (b,h).
// ---------------------------------------------------------------------------
__global__ __launch_bounds__(256) void qkv_project_kernel(
    const float* __restrict__ x,
    const float* __restrict__ Wq, const float* __restrict__ Wk,
    const float* __restrict__ Wv,
    const float* __restrict__ bq, const float* __restrict__ bk,
    const float* __restrict__ bv,
    _Float16* __restrict__ qw, _Float16* __restrict__ kw,
    _Float16* __restrict__ vw)
{
  __shared__ _Float16 Xs[128 * 64];       // 16 KB
  __shared__ _Float16 Ws[3 * 64 * 64];    // 24 KB

  const int bh = blockIdx.x;              // 0..31
  const int b = bh / kH, h = bh % kH;
  const int s0 = blockIdx.y * 128;
  const int tid = threadIdx.x;

  // Stage X tile: float4 global loads, f16 LDS stores.
  for (int i = tid; i < 128 * 16; i += 256) {
    const int r = i >> 4, c4 = (i & 15) * 4;
    const float4 v =
        *(const float4*)&x[(size_t)(b * kS + s0 + r) * kD + h * kDH + c4];
    Xs[r * 64 + c4 + 0] = (_Float16)v.x;
    Xs[r * 64 + c4 + 1] = (_Float16)v.y;
    Xs[r * 64 + c4 + 2] = (_Float16)v.z;
    Xs[r * 64 + c4 + 3] = (_Float16)v.w;
  }
  const float* wsrc[3] = {Wq, Wk, Wv};
#pragma unroll
  for (int m = 0; m < 3; ++m)
    for (int i = tid; i < 1024; i += 256) {
      const float4 v = *(const float4*)&wsrc[m][h * 4096 + i * 4];
      Ws[m * 4096 + i * 4 + 0] = (_Float16)v.x;
      Ws[m * 4096 + i * 4 + 1] = (_Float16)v.y;
      Ws[m * 4096 + i * 4 + 2] = (_Float16)v.z;
      Ws[m * 4096 + i * 4 + 3] = (_Float16)v.w;
    }
  __syncthreads();

  const int wave = tid >> 5, lane = tid & 31;
  const int rA = lane & 15, g = lane >> 4, n = lane & 15;

  // A fragments: 16 rows of X, K split 0..31 / 32..63 (contiguous 16B chunks).
  const _Float16* xrow = &Xs[(wave * 16 + rA) * 64];
  const v16h a0 = frag(*(const v8h*)(xrow + g * 8),
                       *(const v8h*)(xrow + 16 + g * 8));
  const v16h a1 = frag(*(const v8h*)(xrow + 32 + g * 8),
                       *(const v8h*)(xrow + 48 + g * 8));

  const float* bias[3] = {bq, bk, bv};
  _Float16* dst[3] = {qw, kw, vw};

#pragma unroll
  for (int m = 0; m < 3; ++m) {
#pragma unroll
    for (int c = 0; c < 4; ++c) {
      const int o = c * 16 + n;  // output feature = B column
      const _Float16* wrow = &Ws[m * 4096 + o * 64];
      const v16h b0 = frag(*(const v8h*)(wrow + g * 8),
                           *(const v8h*)(wrow + 16 + g * 8));
      const v16h b1 = frag(*(const v8h*)(wrow + 32 + g * 8),
                           *(const v8h*)(wrow + 48 + g * 8));
      v8f acc = {};
      acc = __builtin_amdgcn_wmma_f32_16x16x32_f16(false, a0, false, b0,
                                                   (short)0, acc, false, false);
      acc = __builtin_amdgcn_wmma_f32_16x16x32_f16(false, a1, false, b1,
                                                   (short)0, acc, false, false);
      const float bi = bias[m][h * kDH + o];
      const float sc = (m == 0) ? 0.125f : 1.0f;  // fold 1/sqrt(64) into Q
      _Float16* p = dst[m];
#pragma unroll
      for (int r = 0; r < 8; ++r) {
        const int row = s0 + wave * 16 + r + 8 * g;   // C/D layout row
        const _Float16 hv = (_Float16)((acc[r] + bi) * sc);
        if (m == 2)  // V transposed: [BH][DH][S]
          p[(size_t)(bh * kDH + o) * kS + row] = hv;
        else         // Q,K row-major: [BH][S][DH]
          p[(size_t)(bh * kS + row) * kDH + o] = hv;
      }
    }
  }
}

// ---------------------------------------------------------------------------
// Kernel 2: flash attention. 8 waves share one (b,h); each wave owns 16 query
// rows. K (row-major) and V^T tiles of 32 keys double-buffered in LDS via
// async global->LDS b128 copies.
// ---------------------------------------------------------------------------
__global__ __launch_bounds__(256) void flash_attn_kernel(
    const _Float16* __restrict__ qw, const _Float16* __restrict__ kw,
    const _Float16* __restrict__ vw, float* __restrict__ out)
{
  __shared__ _Float16 Kbuf[2][32 * 64];   // 2 x 4 KB, [key][dim]
  __shared__ _Float16 Vbuf[2][64 * 32];   // 2 x 4 KB, [dim][key] (transposed)
  __shared__ _Float16 Ps[8][16 * 32];     // 8 KB: per-wave P re-layout

  const int bh = blockIdx.x;
  const int b = bh / kH, h = bh % kH;
  const int q0 = blockIdx.y * 128;
  const int tid = threadIdx.x;
  const int wave = tid >> 5, lane = tid & 31;
  const int rA = lane & 15, g = lane >> 4, n = lane & 15;

  const _Float16* kwp = kw + (size_t)bh * kS * kDH;        // [S][DH]
  const _Float16* vwp = vw + (size_t)bh * kDH * kS;        // [DH][S]
  const _Float16* qwp = qw + (size_t)bh * kS * kDH;

  // Q fragments (pre-scaled by 1/8 in kernel 1): contiguous 16B global loads.
  const _Float16* qrow = qwp + (size_t)(q0 + wave * 16 + rA) * kDH;
  const v16h aq0 = frag(*(const v8h*)(qrow + g * 8),
                        *(const v8h*)(qrow + 16 + g * 8));
  const v16h aq1 = frag(*(const v8h*)(qrow + 32 + g * 8),
                        *(const v8h*)(qrow + 48 + g * 8));

  float mrow[8], lrow[8];
  v8f accO[4];
#pragma unroll
  for (int r = 0; r < 8; ++r) { mrow[r] = -1e30f; lrow[r] = 0.0f; }
#pragma unroll
  for (int c = 0; c < 4; ++c)
#pragma unroll
    for (int r = 0; r < 8; ++r) accO[c][r] = 0.0f;

  // Per-thread async-copy addressing (16 bytes per thread per matrix).
  const int vo = tid >> 2, vchunk = tid & 3;               // V^T: row o, chunk
  auto issue_tile = [&](int kt, int buf) {
    // K tile: 32x64 f16 row-major, one contiguous 4 KB block.
    async_b128((unsigned int)(size_t)&Kbuf[buf][0] + tid * 16, kwp,
               (unsigned int)((kt * 2048 + tid * 8) * 2));
    // V^T tile: 64 rows (dims) x 32 keys; row o at vwp + o*S + kt*32.
    async_b128((unsigned int)(size_t)&Vbuf[buf][0] + tid * 16, vwp,
               (unsigned int)(((size_t)vo * kS + kt * 32 + vchunk * 8) * 2));
  };

  issue_tile(0, 0);  // prologue

  for (int kt = 0; kt < kS / 32; ++kt) {
    const int cur = kt & 1;
    wait_async0();       // this wave's copies into buf[cur] are done
    __syncthreads();     // all waves' copies done; prev-iter readers drained
    if (kt + 1 < kS / 32) issue_tile(kt + 1, 1 - cur);

    const _Float16* Kt = Kbuf[cur];
    const _Float16* Vt = Vbuf[cur];

    // Scores: S = Qs @ K^T ; two 16x16 tiles cover 32 keys.
    v8f scv[2];
#pragma unroll
    for (int t = 0; t < 2; ++t) {
      const _Float16* krow = Kt + (t * 16 + n) * 64;  // B column n = key row
      const v16h kb0 = frag(*(const v8h*)(krow + g * 8),
                            *(const v8h*)(krow + 16 + g * 8));
      const v16h kb1 = frag(*(const v8h*)(krow + 32 + g * 8),
                            *(const v8h*)(krow + 48 + g * 8));
      v8f s = {};
      s = __builtin_amdgcn_wmma_f32_16x16x32_f16(false, aq0, false, kb0,
                                                 (short)0, s, false, false);
      s = __builtin_amdgcn_wmma_f32_16x16x32_f16(false, aq1, false, kb1,
                                                 (short)0, s, false, false);
      scv[t] = s;
    }

    // Online softmax (fp32). Row stats reduce across one 16-lane half.
    float p0[8], p1[8], scale[8];
#pragma unroll
    for (int r = 0; r < 8; ++r) {
      float mx = fmaxf(scv[0][r], scv[1][r]);
      mx = fmaxf(mx, __shfl_xor(mx, 1, 16));
      mx = fmaxf(mx, __shfl_xor(mx, 2, 16));
      mx = fmaxf(mx, __shfl_xor(mx, 4, 16));
      mx = fmaxf(mx, __shfl_xor(mx, 8, 16));
      const float mnew = fmaxf(mrow[r], mx);
      scale[r] = __expf(mrow[r] - mnew);
      p0[r] = __expf(scv[0][r] - mnew);
      p1[r] = __expf(scv[1][r] - mnew);
      float rs = p0[r] + p1[r];
      rs += __shfl_xor(rs, 1, 16);
      rs += __shfl_xor(rs, 2, 16);
      rs += __shfl_xor(rs, 4, 16);
      rs += __shfl_xor(rs, 8, 16);
      lrow[r] = lrow[r] * scale[r] + rs;
      mrow[r] = mnew;
    }
#pragma unroll
    for (int c = 0; c < 4; ++c)
#pragma unroll
      for (int r = 0; r < 8; ++r) accO[c][r] *= scale[r];

    // Re-layout P (C/D -> A operand) through wave-private LDS.
    // Same-wave LDS ops are processed in order, so the gather sees the stores.
#pragma unroll
    for (int r = 0; r < 8; ++r) {
      const int row = r + 8 * g;
      Ps[wave][row * 32 + n]      = (_Float16)p0[r];
      Ps[wave][row * 32 + 16 + n] = (_Float16)p1[r];
    }
    const _Float16* prow = &Ps[wave][rA * 32];
    const v16h ap = frag(*(const v8h*)(prow + g * 8),
                         *(const v8h*)(prow + 16 + g * 8));

    // O += P @ V : K-dim = 32 keys; V^T rows are contiguous.
#pragma unroll
    for (int c = 0; c < 4; ++c) {
      const _Float16* vrow = Vt + (c * 16 + n) * 32;  // B column n: dim row
      const v16h vb = frag(*(const v8h*)(vrow + g * 8),
                           *(const v8h*)(vrow + 16 + g * 8));
      accO[c] = __builtin_amdgcn_wmma_f32_16x16x32_f16(false, ap, false, vb,
                                                       (short)0, accO[c],
                                                       false, false);
    }
    // No trailing barrier: next iteration's wait+barrier separates buffer use.
  }

  // Normalize and write out[b, q, h*64 + o] in fp32.
#pragma unroll
  for (int r = 0; r < 8; ++r) {
    const float inv = 1.0f / lrow[r];
    const int row = q0 + wave * 16 + r + 8 * g;
#pragma unroll
    for (int c = 0; c < 4; ++c)
      out[(size_t)(b * kS + row) * kD + h * kDH + c * 16 + n] =
          accO[c][r] * inv;
  }
}

// ---------------------------------------------------------------------------
extern "C" void kernel_launch(void* const* d_in, const int* in_sizes, int n_in,
                              void* d_out, int out_size, void* d_ws,
                              size_t ws_size, hipStream_t stream) {
  const float* x  = (const float*)d_in[0];
  const float* Wq = (const float*)d_in[1];
  const float* Wk = (const float*)d_in[2];
  const float* Wv = (const float*)d_in[3];
  const float* bq = (const float*)d_in[4];
  const float* bk = (const float*)d_in[5];
  const float* bv = (const float*)d_in[6];
  float* out = (float*)d_out;

  const size_t per = (size_t)kB * kH * kS * kDH;  // 4,194,304 elems
  _Float16* qw = (_Float16*)d_ws;
  _Float16* kw = qw + per;
  _Float16* vw = kw + per;

  dim3 grid(kB * kH, kS / 128);  // 32 x 16 blocks, 8 waves each
  qkv_project_kernel<<<grid, 256, 0, stream>>>(x, Wq, Wk, Wv, bq, bk, bv,
                                               qw, kw, vw);
  flash_attn_kernel<<<grid, 256, 0, stream>>>(qw, kw, vw, out);
}
